// FineGrainedCrossModalAttention_18837726560682
// MI455X (gfx1250) — matrix operand order repaired and checked by reading, not compiled, not run on hardware
//
#include <hip/hip_runtime.h>
#include <hip/hip_bf16.h>

// ---------------------------------------------------------------------------
// FineGrainedCrossModalAttention for MI455X (gfx1250, wave32, WMMA).
// Pipeline (all f16 intermediates, f32 accumulation):
//   1. convert activations + weights f32->f16
//   2. WMMA GEMMs for projections / QKV (V written pre-transposed per head)
//   3. flash-attention kernel (online softmax, WMMA QK^T and PV)
//   4. WMMA GEMMs for output linears (f32 out)
//   5. residual + LayerNorm
// Workspace requirement: ~154 MB.
// ---------------------------------------------------------------------------

typedef __attribute__((ext_vector_type(16))) _Float16 v16h;
typedef __attribute__((ext_vector_type(8)))  _Float16 v8h;
typedef __attribute__((ext_vector_type(4)))  _Float16 v4h;
typedef __attribute__((ext_vector_type(8)))  float    v8f;

union V16U { v16h v; v8h h[2]; };

#define WMMA_F16(a, b, c) \
  __builtin_amdgcn_wmma_f32_16x16x32_f16(false, (a), false, (b), (short)0, (c), false, false)

#define BATCH   32
#define NATOMS  256
#define SEQLEN  512
#define NODED   256
#define TOKD    768
#define HID     256
#define NHEADS  8
#define HDIM    32

static constexpr float ATTN_SCALE = 0.17677669529663688f; // 32^-0.5

// ---------------------------------------------------------------------------
// f32 -> f16 conversion, 4 elements / thread
// ---------------------------------------------------------------------------
__global__ void cvt_f32_to_f16_x4(const float* __restrict__ in,
                                  _Float16* __restrict__ out, int n4) {
  int i = blockIdx.x * blockDim.x + threadIdx.x;
  if (i >= n4) return;
  float4 f = ((const float4*)in)[i];
  v4h o;
  o[0] = (_Float16)f.x; o[1] = (_Float16)f.y;
  o[2] = (_Float16)f.z; o[3] = (_Float16)f.w;
  ((v4h*)out)[i] = o;
}

// ---------------------------------------------------------------------------
// WMMA fragment loads.
// A frag (16x32 f16, M x K): lane (lr,hi) holds row lr; halves at
//   K = k0 + hi*16 .. +7  and  K = k0 + hi*16 + 8 .. +7   (two 16B chunks)
// B frag (32x16 f16, K x N): lane (lr,hi) holds column lr; 16 contiguous
//   halves at K = k0 + hi*16   (one 32B chunk)
// ---------------------------------------------------------------------------
__device__ __forceinline__ v16h load_a_frag(const _Float16* A, int K, int row,
                                            int k0, int hi) {
  const _Float16* p = A + (size_t)row * K + k0 + hi * 16;
  V16U u;
  u.h[0] = *(const v8h*)p;
  u.h[1] = *(const v8h*)(p + 8);
  return u.v;
}

__device__ __forceinline__ v16h load_b_frag(const _Float16* W, int K, int col,
                                            int k0, int hi) {
  return *(const v16h*)(W + (size_t)col * K + k0 + hi * 16);
}

// C/D layout: VGPR r, lanes 0-15 -> row r, lanes 16-31 -> row r+8; col = lane&15
__device__ __forceinline__ void store_tile(v8f c, int mtop, int col, float bv,
                                           float* __restrict__ o32,
                                           _Float16* __restrict__ o16,
                                           int N, int mode, int seqlen, int hi) {
#pragma unroll
  for (int r = 0; r < 8; ++r) {
    int row = mtop + hi * 8 + r;
    float v = c[r] + bv;
    if (mode == 0) {                       // f32 row-major
      o32[(size_t)row * N + col] = v;
    } else if (mode == 1) {                // f16 row-major
      o16[(size_t)row * N + col] = (_Float16)v;
    } else {                               // f16 V^T: [b, h, d, m]
      int bb = row / seqlen, m = row - bb * seqlen;
      int h = col >> 5, d = col & 31;
      o16[(((size_t)bb * NHEADS + h) * HDIM + d) * (size_t)seqlen + m] = (_Float16)v;
    }
  }
}

// ---------------------------------------------------------------------------
// GEMM: out[M,N] = A[M,K](f16) * W[N,K](f16)^T + bias[N]
// One wave computes a 32x32 tile (2x2 WMMA C-tiles), K-loop step 32.
// ---------------------------------------------------------------------------
__global__ void gemm_wmma(const _Float16* __restrict__ A,
                          const _Float16* __restrict__ W,
                          const float* __restrict__ bias,
                          float* __restrict__ o32, _Float16* __restrict__ o16,
                          int M, int N, int K, int mode, int seqlen) {
  int lane = threadIdx.x & 31;
  int wid  = (blockIdx.x * blockDim.x + threadIdx.x) >> 5;
  int tilesM = M >> 5, tilesN = N >> 5;
  if (wid >= tilesM * tilesN) return;
  int tm = wid % tilesM, tn = wid / tilesM;
  int m0 = tm << 5, n0 = tn << 5;
  int lr = lane & 15, hi = lane >> 4;

  v8f c00 = {}, c01 = {}, c10 = {}, c11 = {};
  for (int k0 = 0; k0 < K; k0 += 32) {
    v16h a0 = load_a_frag(A, K, m0 + lr,      k0, hi);
    v16h a1 = load_a_frag(A, K, m0 + 16 + lr, k0, hi);
    v16h b0 = load_b_frag(W, K, n0 + lr,      k0, hi);
    v16h b1 = load_b_frag(W, K, n0 + 16 + lr, k0, hi);
    __builtin_prefetch(A + (size_t)(m0 + lr) * K + k0 + 32, 0, 0);
    c00 = WMMA_F16(a0, b0, c00);
    c01 = WMMA_F16(a0, b1, c01);
    c10 = WMMA_F16(a1, b0, c10);
    c11 = WMMA_F16(a1, b1, c11);
  }
  float bv0 = bias[n0 + lr];
  float bv1 = bias[n0 + 16 + lr];
  store_tile(c00, m0,      n0 + lr,      bv0, o32, o16, N, mode, seqlen, hi);
  store_tile(c01, m0,      n0 + 16 + lr, bv1, o32, o16, N, mode, seqlen, hi);
  store_tile(c10, m0 + 16, n0 + lr,      bv0, o32, o16, N, mode, seqlen, hi);
  store_tile(c11, m0 + 16, n0 + 16 + lr, bv1, o32, o16, N, mode, seqlen, hi);
}

// ---------------------------------------------------------------------------
// Flash cross-attention. One wave owns (b, h, 16-query tile); streams keys
// in chunks of 32 with online softmax. Q/K are [b*rows, HID] f16 (head h in
// cols h*32..h*32+31); VT is [b, h, d, m] f16; ctx out is [b*rows, HID] f16.
// ---------------------------------------------------------------------------
__global__ void flash_attn(const _Float16* __restrict__ Q,
                           const _Float16* __restrict__ Km,
                           const _Float16* __restrict__ VT,
                           const unsigned char* __restrict__ mask,
                           _Float16* __restrict__ ctx, int Nq, int Mk) {
  __shared__ __align__(32) _Float16 pstage[8][16 * 32];  // per-wave P tile
  int lane = threadIdx.x & 31;
  int wloc = threadIdx.x >> 5;
  int gwid = blockIdx.x * 8 + wloc;
  int qt = Nq >> 4;
  if (gwid >= BATCH * NHEADS * qt) return;
  int b   = gwid / (NHEADS * qt);
  int rem = gwid % (NHEADS * qt);
  int h   = rem / qt;
  int q0  = (rem % qt) << 4;
  int lr = lane & 15, hi = lane >> 4;
  _Float16* pst = pstage[wloc];

  // Q fragment (16 queries x 32 head dims) — exactly one WMMA K-step
  const _Float16* qp = Q + ((size_t)b * Nq + q0 + lr) * HID + h * HDIM + hi * 16;
  V16U qa;
  qa.h[0] = *(const v8h*)qp;
  qa.h[1] = *(const v8h*)(qp + 8);
  v16h qf = qa.v;

  float mrow[8], lrow[8];
#pragma unroll
  for (int r = 0; r < 8; ++r) { mrow[r] = -3.0e38f; lrow[r] = 0.0f; }
  v8f acc0 = {}, acc1 = {};
  const v8f zero = {};
  const _Float16* vbase = VT + ((size_t)b * NHEADS + h) * HDIM * (size_t)Mk;

  for (int kc = 0; kc < Mk; kc += 32) {
    // scores for 32 keys: two 16-key B fragments
    v16h kf0 = *(const v16h*)(Km + ((size_t)b * Mk + kc + lr) * HID + h * HDIM + hi * 16);
    v16h kf1 = *(const v16h*)(Km + ((size_t)b * Mk + kc + 16 + lr) * HID + h * HDIM + hi * 16);
    v8f s0 = WMMA_F16(qf, kf0, zero);
    v8f s1 = WMMA_F16(qf, kf1, zero);

    float msk0 = mask[(size_t)b * Mk + kc + lr]      ? 0.0f : -1.0e30f;
    float msk1 = mask[(size_t)b * Mk + kc + 16 + lr] ? 0.0f : -1.0e30f;

#pragma unroll
    for (int r = 0; r < 8; ++r) {
      float e0 = s0[r] * ATTN_SCALE + msk0;
      float e1 = s1[r] * ATTN_SCALE + msk1;
      // row-max across the 16 lanes of this half-wave
      float tmax = fmaxf(e0, e1);
      for (int o = 1; o < 16; o <<= 1) tmax = fmaxf(tmax, __shfl_xor(tmax, o, 32));
      float mnew = fmaxf(mrow[r], tmax);
      float sf = __expf(mrow[r] - mnew);
      float p0 = __expf(e0 - mnew);
      float p1 = __expf(e1 - mnew);
      float ps = p0 + p1;
      for (int o = 1; o < 16; o <<= 1) ps += __shfl_xor(ps, o, 32);
      lrow[r] = lrow[r] * sf + ps;
      mrow[r] = mnew;
      acc0[r] *= sf;
      acc1[r] *= sf;
      // stage P (C layout -> LDS row-major 16x32)
      int prow = hi * 8 + r;
      pst[prow * 32 + lr]      = (_Float16)p0;
      pst[prow * 32 + 16 + lr] = (_Float16)p1;
    }
    asm volatile("s_wait_dscnt 0" ::: "memory");  // P tile RAW (cross-lane)

    // reload P as an A fragment (16 queries x 32 keys)
    V16U pa;
    pa.h[0] = *(const v8h*)&pst[lr * 32 + hi * 16];
    pa.h[1] = *(const v8h*)&pst[lr * 32 + hi * 16 + 8];

    // V^T fragments: column = output dim d, K = 32 keys (contiguous in VT)
    v16h vf0 = *(const v16h*)(vbase + (size_t)lr * Mk + kc + hi * 16);
    v16h vf1 = *(const v16h*)(vbase + (size_t)(16 + lr) * Mk + kc + hi * 16);
    acc0 = WMMA_F16(pa.v, vf0, acc0);
    acc1 = WMMA_F16(pa.v, vf1, acc1);
  }

#pragma unroll
  for (int r = 0; r < 8; ++r) {
    float inv = 1.0f / lrow[r];
    size_t row = (size_t)b * Nq + q0 + hi * 8 + r;
    ctx[row * HID + h * HDIM + lr]      = (_Float16)(acc0[r] * inv);
    ctx[row * HID + h * HDIM + 16 + lr] = (_Float16)(acc1[r] * inv);
  }
}

// ---------------------------------------------------------------------------
// Residual + LayerNorm: out = LN(orig + delta) * g + beta.  One block / row.
// ---------------------------------------------------------------------------
__global__ void residual_layernorm(const float* __restrict__ orig,
                                   const float* __restrict__ delta,
                                   const float* __restrict__ g,
                                   const float* __restrict__ beta,
                                   float* __restrict__ out, int D) {
  int row = blockIdx.x;
  int tid = threadIdx.x;
  int per = D >> 8;  // D is 256 or 768 -> 1 or 3 per thread
  float x[3];
  float s = 0.0f, ss = 0.0f;
  const float* po = orig + (size_t)row * D;
  const float* pd = delta + (size_t)row * D;
  for (int i = 0; i < per; ++i) {
    int idx = tid + (i << 8);
    float v = po[idx] + pd[idx];
    x[i] = v;
    s += v;
    ss += v * v;
  }
  for (int o = 1; o < 32; o <<= 1) {
    s += __shfl_xor(s, o, 32);
    ss += __shfl_xor(ss, o, 32);
  }
  __shared__ float sh[16];
  if ((tid & 31) == 0) {
    sh[tid >> 5] = s;
    sh[8 + (tid >> 5)] = ss;
  }
  __syncthreads();
  float ts = 0.0f, tss = 0.0f;
  for (int w = 0; w < 8; ++w) { ts += sh[w]; tss += sh[8 + w]; }
  float mu = ts / D;
  float var = tss / D - mu * mu;
  float rs = rsqrtf(var + 1e-5f);
  float* pw = out + (size_t)row * D;
  for (int i = 0; i < per; ++i) {
    int idx = tid + (i << 8);
    pw[idx] = (x[i] - mu) * rs * g[idx] + beta[idx];
  }
}

// ---------------------------------------------------------------------------
// Host orchestration
// ---------------------------------------------------------------------------
extern "C" void kernel_launch(void* const* d_in, const int* in_sizes, int n_in,
                              void* d_out, int out_size, void* d_ws, size_t ws_size,
                              hipStream_t stream) {
  const float* node_feat   = (const float*)d_in[0];
  const float* token_feat  = (const float*)d_in[1];
  const unsigned char* node_mask  = (const unsigned char*)d_in[2];
  const unsigned char* token_mask = (const unsigned char*)d_in[3];
  const float* node_proj_w  = (const float*)d_in[4];
  const float* node_proj_b  = (const float*)d_in[5];
  const float* token_proj_w = (const float*)d_in[6];
  const float* token_proj_b = (const float*)d_in[7];
  const float* a2t_q_w = (const float*)d_in[8];
  const float* a2t_q_b = (const float*)d_in[9];
  const float* a2t_k_w = (const float*)d_in[10];
  const float* a2t_k_b = (const float*)d_in[11];
  const float* a2t_v_w = (const float*)d_in[12];
  const float* a2t_v_b = (const float*)d_in[13];
  const float* t2a_q_w = (const float*)d_in[14];
  const float* t2a_q_b = (const float*)d_in[15];
  const float* t2a_k_w = (const float*)d_in[16];
  const float* t2a_k_b = (const float*)d_in[17];
  const float* t2a_v_w = (const float*)d_in[18];
  const float* t2a_v_b = (const float*)d_in[19];
  const float* node_out_w  = (const float*)d_in[20];
  const float* node_out_b  = (const float*)d_in[21];
  const float* token_out_w = (const float*)d_in[22];
  const float* token_out_b = (const float*)d_in[23];
  const float* ln_node_g  = (const float*)d_in[24];
  const float* ln_node_b  = (const float*)d_in[25];
  const float* ln_token_g = (const float*)d_in[26];
  const float* ln_token_b = (const float*)d_in[27];

  const int MN = BATCH * NATOMS;  // 8192 node rows
  const int MT = BATCH * SEQLEN;  // 16384 token rows

  char* ws = (char*)d_ws;
  size_t off = 0;
  auto alloc = [&](size_t bytes) -> void* {
    void* p = ws + off;
    off = (off + bytes + 255) & ~(size_t)255;
    return p;
  };

  _Float16* nf16    = (_Float16*)alloc((size_t)MN * NODED * 2);
  _Float16* tf16    = (_Float16*)alloc((size_t)MT * TOKD * 2);
  _Float16* w_np16  = (_Float16*)alloc((size_t)HID * NODED * 2);
  _Float16* w_tp16  = (_Float16*)alloc((size_t)HID * TOKD * 2);
  _Float16* w_aq16  = (_Float16*)alloc((size_t)HID * HID * 2);
  _Float16* w_ak16  = (_Float16*)alloc((size_t)HID * HID * 2);
  _Float16* w_av16  = (_Float16*)alloc((size_t)HID * HID * 2);
  _Float16* w_tq16  = (_Float16*)alloc((size_t)HID * HID * 2);
  _Float16* w_tk16  = (_Float16*)alloc((size_t)HID * HID * 2);
  _Float16* w_tv16  = (_Float16*)alloc((size_t)HID * HID * 2);
  _Float16* w_no16  = (_Float16*)alloc((size_t)NODED * HID * 2);
  _Float16* w_to16  = (_Float16*)alloc((size_t)TOKD * HID * 2);
  _Float16* nproj16 = (_Float16*)alloc((size_t)MN * HID * 2);
  _Float16* tproj16 = (_Float16*)alloc((size_t)MT * HID * 2);
  _Float16* q_a2t   = (_Float16*)alloc((size_t)MN * HID * 2);
  _Float16* k_a2t   = (_Float16*)alloc((size_t)MT * HID * 2);
  _Float16* vt_a2t  = (_Float16*)alloc((size_t)MT * HID * 2);  // [b,h,d,m]
  _Float16* q_t2a   = (_Float16*)alloc((size_t)MT * HID * 2);
  _Float16* k_t2a   = (_Float16*)alloc((size_t)MN * HID * 2);
  _Float16* vt_t2a  = (_Float16*)alloc((size_t)MN * HID * 2);  // [b,h,d,m]
  _Float16* ctx_a   = (_Float16*)alloc((size_t)MN * HID * 2);
  _Float16* ctx_t   = (_Float16*)alloc((size_t)MT * HID * 2);
  float* dnode      = (float*)alloc((size_t)MN * NODED * 4);
  float* dtok       = (float*)alloc((size_t)MT * TOKD * 4);
  (void)ws_size;

  auto cvt = [&](const float* src, _Float16* dst, size_t n) {
    int n4 = (int)(n >> 2);
    cvt_f32_to_f16_x4<<<(n4 + 255) / 256, 256, 0, stream>>>(src, dst, n4);
  };
  auto gemm = [&](const _Float16* A, const _Float16* W, const float* bias,
                  void* out, int M, int N, int K, int mode, int seqlen) {
    int waves = (M >> 5) * (N >> 5);
    gemm_wmma<<<(waves + 7) / 8, 256, 0, stream>>>(
        A, W, bias, (float*)out, (_Float16*)out, M, N, K, mode, seqlen);
  };

  // 1. conversions
  cvt(node_feat, nf16, (size_t)MN * NODED);
  cvt(token_feat, tf16, (size_t)MT * TOKD);
  cvt(node_proj_w, w_np16, (size_t)HID * NODED);
  cvt(token_proj_w, w_tp16, (size_t)HID * TOKD);
  cvt(a2t_q_w, w_aq16, (size_t)HID * HID);
  cvt(a2t_k_w, w_ak16, (size_t)HID * HID);
  cvt(a2t_v_w, w_av16, (size_t)HID * HID);
  cvt(t2a_q_w, w_tq16, (size_t)HID * HID);
  cvt(t2a_k_w, w_tk16, (size_t)HID * HID);
  cvt(t2a_v_w, w_tv16, (size_t)HID * HID);
  cvt(node_out_w, w_no16, (size_t)NODED * HID);
  cvt(token_out_w, w_to16, (size_t)TOKD * HID);

  // 2. input projections
  gemm(nf16, w_np16, node_proj_b, nproj16, MN, HID, NODED, 1, 0);
  gemm(tf16, w_tp16, token_proj_b, tproj16, MT, HID, TOKD, 1, 0);

  // 3. atom->text attention
  gemm(nproj16, w_aq16, a2t_q_b, q_a2t, MN, HID, HID, 1, 0);
  gemm(tproj16, w_ak16, a2t_k_b, k_a2t, MT, HID, HID, 1, 0);
  gemm(tproj16, w_av16, a2t_v_b, vt_a2t, MT, HID, HID, 2, SEQLEN);
  {
    int waves = BATCH * NHEADS * (NATOMS / 16);
    flash_attn<<<waves / 8, 256, 0, stream>>>(q_a2t, k_a2t, vt_a2t, token_mask,
                                              ctx_a, NATOMS, SEQLEN);
  }
  gemm(ctx_a, w_no16, node_out_b, dnode, MN, NODED, HID, 0, 0);

  // 4. text->atom attention
  gemm(tproj16, w_tq16, t2a_q_b, q_t2a, MT, HID, HID, 1, 0);
  gemm(nproj16, w_tk16, t2a_k_b, k_t2a, MN, HID, HID, 1, 0);
  gemm(nproj16, w_tv16, t2a_v_b, vt_t2a, MN, HID, HID, 2, NATOMS);
  {
    int waves = BATCH * NHEADS * (SEQLEN / 16);
    flash_attn<<<waves / 8, 256, 0, stream>>>(q_t2a, k_t2a, vt_t2a, node_mask,
                                              ctx_t, SEQLEN, NATOMS);
  }
  gemm(ctx_t, w_to16, token_out_b, dtok, MT, TOKD, HID, 0, 0);

  // 5. residual + LayerNorm into d_out (nodes first, then tokens)
  float* out_nodes = (float*)d_out;
  float* out_tokens = out_nodes + (size_t)MN * NODED;
  residual_layernorm<<<MN, 256, 0, stream>>>(node_feat, dnode, ln_node_g,
                                             ln_node_b, out_nodes, NODED);
  residual_layernorm<<<MT, 256, 0, stream>>>(token_feat, dtok, ln_token_g,
                                             ln_token_b, out_tokens, TOKD);
}
